// Attention_15539191677786
// MI455X (gfx1250) — matrix operand order repaired and checked by reading, not compiled, not run on hardware
//
#include <hip/hip_runtime.h>
#include <hip/hip_bf16.h>
#include <hip/hip_fp16.h>
#include <math.h>
#include <stdint.h>

// ---------------- problem constants ----------------
constexpr int NN  = 2048;   // sequence length
constexpr int DD  = 1024;   // model dim
constexpr int NH  = 16;     // heads
constexpr int HWC = 64;     // head width
constexpr float EPSV = 1e-5f;

typedef __attribute__((ext_vector_type(16))) _Float16 v16h;
typedef __attribute__((ext_vector_type(8)))  float    v8f;
typedef __attribute__((ext_vector_type(4)))  unsigned int u32x4;
typedef __attribute__((ext_vector_type(4)))  int      i32x4;
typedef __attribute__((ext_vector_type(8)))  int      i32x8;

union Frag  { v16h v; int4 q[2]; };
union F16x8 { int4 i; _Float16 h[8]; };

// ---------------- TDM availability / arity detection ----------------
#if defined(__has_builtin)
#  if __has_builtin(__builtin_amdgcn_tensor_load_to_lds)
#    define HAS_TDM 1
#  endif
#endif
#ifndef HAS_TDM
#  define HAS_TDM 0
#endif
#if defined(__has_include)
#  if __has_include(<hip/amd_detail/amd_gfx1250_TDM.h>)
#    define TDM_6ARG 1   // therock-10.0 headers -> clang-23 6-arg builtin
#  endif
#endif

#if HAS_TDM
// 2-D f16 tile DMA: global (row-major, row_stride elems) -> LDS, with optional
// LDS row padding (pad codes per D# group1: interval 2^(c+1) DW, amount c+1 DW).
__device__ __forceinline__ void tdm_load_2d_f16(
    unsigned lds_byte_addr, const void* gaddr,
    unsigned tile_w_elems, unsigned tile_h_rows,
    unsigned long long row_stride_elems,
    int pad_interval_code, int pad_amount_code) {
  const unsigned long long ga = (unsigned long long)(uintptr_t)gaddr;
  // D# group0: count=1 (valid user descriptor), lds_addr, 57-bit global addr, type=2
  u32x4 g0 = { 1u,
               lds_byte_addr,
               (unsigned)(ga & 0xFFFFFFFFull),
               (unsigned)((ga >> 32) & 0x01FFFFFFull) | 0x80000000u };
  unsigned w0 = (1u << 16);  // data_size = 1 -> 2 bytes; wg_mask = 0 (not in cluster)
  if (pad_interval_code >= 0)
    w0 |= (1u << 20) | ((unsigned)pad_interval_code << 22)
        | ((unsigned)pad_amount_code << 25);
  const unsigned td0 = tile_w_elems, td1 = tile_h_rows;
  // group1 bit packing: tensor_dim0[79:48], tensor_dim1[111:80], tile_dim0[127:112],
  // tile_dim1[143:128], tensor_dim0_stride[207:160]
  i32x8 g1 = { (int)w0,
               (int)((td0 & 0xFFFFu) << 16),
               (int)(((td0 >> 16) & 0xFFFFu) | ((td1 & 0xFFFFu) << 16)),
               (int)(((td1 >> 16) & 0xFFFFu) | ((tile_w_elems & 0xFFFFu) << 16)),
               (int)(tile_h_rows & 0xFFFFu),                  // tile_dim1; tile_dim2=0
               (int)(row_stride_elems & 0xFFFFFFFFull),
               (int)((row_stride_elems >> 32) & 0xFFFFull),   // dim1_stride unused (2D)
               0 };
  i32x4 gz = {0, 0, 0, 0};
#ifdef TDM_6ARG
  i32x8 gz8 = {0, 0, 0, 0, 0, 0, 0, 0};
  __builtin_amdgcn_tensor_load_to_lds(g0, g1, gz, gz, gz8, 0);
#else
  __builtin_amdgcn_tensor_load_to_lds(g0, g1, gz, gz, 0);
#endif
}
#endif  // HAS_TDM

__device__ __forceinline__ v8f wmma_f16(v16h a, v16h b, v8f c) {
  return __builtin_amdgcn_wmma_f32_16x16x32_f16(false, a, false, b, (short)0, c,
                                                false, false);
}

__device__ __forceinline__ float redmax16(float v) {
#pragma unroll
  for (int i = 1; i < 16; i <<= 1) v = fmaxf(v, __shfl_xor(v, i, 32));
  return v;
}
__device__ __forceinline__ float redsum16(float v) {
#pragma unroll
  for (int i = 1; i < 16; i <<= 1) v += __shfl_xor(v, i, 32);
  return v;
}

// ---------------- f32 -> f16 convert ----------------
__global__ void cvt_kernel(const float* __restrict__ src, _Float16* __restrict__ dst, int n) {
  for (int i = blockIdx.x * blockDim.x + threadIdx.x; i < n; i += gridDim.x * blockDim.x)
    dst[i] = (_Float16)src[i];
}

// ---------------- LayerNorm: x (N,D) f32 -> xn f16 ----------------
__global__ __launch_bounds__(256) void ln_kernel(const float* __restrict__ x,
                                                 const float* __restrict__ w,
                                                 const float* __restrict__ b,
                                                 _Float16* __restrict__ xn) {
  const int row = blockIdx.x;
  const int t = threadIdx.x;
  const float4 xv = *(const float4*)(x + (size_t)row * DD + t * 4);
  float s  = xv.x + xv.y + xv.z + xv.w;
  float ss = xv.x * xv.x + xv.y * xv.y + xv.z * xv.z + xv.w * xv.w;
#pragma unroll
  for (int i = 1; i < 32; i <<= 1) { s += __shfl_xor(s, i, 32); ss += __shfl_xor(ss, i, 32); }
  __shared__ float rs[8], rss[8];
  const int wave = t >> 5, lane = t & 31;
  if (lane == 0) { rs[wave] = s; rss[wave] = ss; }
  __syncthreads();
  if (wave == 0) {
    float a  = (lane < 8) ? rs[lane]  : 0.f;
    float a2 = (lane < 8) ? rss[lane] : 0.f;
#pragma unroll
    for (int i = 1; i < 8; i <<= 1) { a += __shfl_xor(a, i, 32); a2 += __shfl_xor(a2, i, 32); }
    if (lane == 0) { rs[0] = a; rss[0] = a2; }
  }
  __syncthreads();
  const float mu  = rs[0] * (1.0f / DD);
  const float var = rss[0] * (1.0f / DD) - mu * mu;
  const float inv = rsqrtf(var + EPSV);
  const float xe[4] = {xv.x, xv.y, xv.z, xv.w};
#pragma unroll
  for (int j = 0; j < 4; j++) {
    const int i = t * 4 + j;
    xn[(size_t)row * DD + i] = (_Float16)((xe[j] - mu) * inv * w[i] + b[i]);
  }
}

// ---------------- generic WMMA GEMM: out = A(N x K) * B(M x K)^T ----------------
// 256 threads (8 waves); tile 128x128; K step 32; TDM double-buffered staging.
// EPI: 0 = QKV scatter, 1 = gate sigmoid(+bg)*y, 2 = +bo -> f32
template <int EPI>
__global__ __launch_bounds__(256) void gemm_kernel(
    const _Float16* __restrict__ A, const _Float16* __restrict__ B, int K,
    const float* __restrict__ bias, const _Float16* __restrict__ aux,
    _Float16* __restrict__ outh, float* __restrict__ outf,
    _Float16* __restrict__ qo, _Float16* __restrict__ ko, _Float16* __restrict__ vo) {
  constexpr int LDT = 40;  // 32 + 8 pad (f16) -> pad codes: interval 16DW=3, amount 4DW=3
  __shared__ __align__(16) _Float16 As[2][128 * LDT];
  __shared__ __align__(16) _Float16 Bs[2][128 * LDT];

  const int t = threadIdx.x;
  const int wave = t >> 5, lane = t & 31, lm = lane & 15, hh = lane >> 4;
  const int nBase = blockIdx.y * 128, mBase = blockIdx.x * 128;
  const int rowOff = (wave & 1) * 64, colOff = (wave >> 1) * 32;
  const int nk = K / 32;

  v8f acc[4][2];
#pragma unroll
  for (int s = 0; s < 4; s++)
#pragma unroll
    for (int c = 0; c < 2; c++) acc[s][c] = (v8f){};

#if HAS_TDM
  if (wave == 0) {
    tdm_load_2d_f16((unsigned)(uintptr_t)&As[0][0], &A[(size_t)nBase * K], 32, 128,
                    (unsigned long long)K, 3, 3);
    tdm_load_2d_f16((unsigned)(uintptr_t)&Bs[0][0], &B[(size_t)mBase * K], 32, 128,
                    (unsigned long long)K, 3, 3);
    __builtin_amdgcn_s_wait_tensorcnt(0);
  }
  __syncthreads();
#endif

  for (int it = 0; it < nk; ++it) {
#if HAS_TDM
    const int p = it & 1;
    if (wave == 0 && it + 1 < nk) {  // prefetch next tiles into other buffer
      const int k0n = (it + 1) * 32;
      tdm_load_2d_f16((unsigned)(uintptr_t)&As[p ^ 1][0],
                      &A[(size_t)nBase * K + k0n], 32, 128, (unsigned long long)K, 3, 3);
      tdm_load_2d_f16((unsigned)(uintptr_t)&Bs[p ^ 1][0],
                      &B[(size_t)mBase * K + k0n], 32, 128, (unsigned long long)K, 3, 3);
    }
#else
    const int p = 0;
    const int k0 = it * 32;
#pragma unroll
    for (int rnd = 0; rnd < 2; rnd++) {
      const int idx = t + rnd * 256;
      const int r = idx >> 2, cg = idx & 3;
      *(int4*)(&As[0][r * LDT + cg * 8]) =
          *(const int4*)(&A[(size_t)(nBase + r) * K + k0 + cg * 8]);
      *(int4*)(&Bs[0][r * LDT + cg * 8]) =
          *(const int4*)(&B[(size_t)(mBase + r) * K + k0 + cg * 8]);
    }
    __syncthreads();
#endif

    Frag af[4], bf[2];
#pragma unroll
    for (int s = 0; s < 4; s++) {
      const int rr = (rowOff + s * 16 + lm) * LDT;
      af[s].q[0] = *(const int4*)(&As[p][rr + hh * 8]);
      af[s].q[1] = *(const int4*)(&As[p][rr + 16 + hh * 8]);
    }
#pragma unroll
    for (int c = 0; c < 2; c++) {
      const int rr = (colOff + c * 16 + lm) * LDT;
      bf[c].q[0] = *(const int4*)(&Bs[p][rr + hh * 8]);
      bf[c].q[1] = *(const int4*)(&Bs[p][rr + 16 + hh * 8]);
    }
#pragma unroll
    for (int s = 0; s < 4; s++)
#pragma unroll
      for (int c = 0; c < 2; c++) acc[s][c] = wmma_f16(af[s].v, bf[c].v, acc[s][c]);

#if HAS_TDM
    if (wave == 0) __builtin_amdgcn_s_wait_tensorcnt(0);
#endif
    __syncthreads();
  }

  // epilogue: C layout -> lane lm = col, element r -> row r + 8*hh
#pragma unroll
  for (int s = 0; s < 4; s++)
#pragma unroll
    for (int c = 0; c < 2; c++)
#pragma unroll
      for (int r = 0; r < 8; r++) {
        const int row = nBase + rowOff + s * 16 + r + 8 * hh;
        const int col = mBase + colOff + c * 16 + lm;
        const float v = acc[s][c][r];
        if (EPI == 0) {
          const int h = col / 192, j = col % 192, kind = j / 64, cc = j % 64;
          const size_t o = ((size_t)h * NN + row) * HWC + cc;
          if (kind == 0)      qo[o] = (_Float16)(v * 0.125f);  // HW^-0.5
          else if (kind == 1) ko[o] = (_Float16)v;
          else                vo[o] = (_Float16)v;
        } else if (EPI == 1) {
          const float g = 1.0f / (1.0f + __expf(-(v + bias[col])));
          outh[(size_t)row * DD + col] =
              (_Float16)(g * (float)aux[(size_t)row * DD + col]);
        } else {
          outf[(size_t)row * DD + col] = v + bias[col];
        }
      }
}

// ---------------- flash attention: 1 head x 128 query rows per block ----------------
__global__ __launch_bounds__(256) void attn_kernel(
    const _Float16* __restrict__ qh, const _Float16* __restrict__ kh,
    const _Float16* __restrict__ vh, const float* __restrict__ bias,
    const int* __restrict__ mask, _Float16* __restrict__ yh) {
  constexpr int LQ = 72, LK = 72, LV = 40, LP = 40;  // pad codes: 32DW rows -> 4 / 3
  __shared__ __align__(16) _Float16 Qs[128 * LQ];
  __shared__ __align__(16) _Float16 Ks[32 * LK];
  __shared__ __align__(16) _Float16 VTs[64 * LV];   // transposed V: [c][key]
  __shared__ __align__(16) _Float16 Pw[8][16 * LP]; // per-wave P relayout scratch

  const int t = threadIdx.x, wave = t >> 5, lane = t & 31, lm = lane & 15, hh = lane >> 4;
  const int h = blockIdx.y, qBlk = blockIdx.x * 128;

  // stage Q tile (128 x 64 f16)
#if HAS_TDM
  if (wave == 0) {
    tdm_load_2d_f16((unsigned)(uintptr_t)Qs, &qh[((size_t)h * NN + qBlk) * HWC],
                    64, 128, 64ull, 4, 3);
    __builtin_amdgcn_s_wait_tensorcnt(0);
  }
#else
#pragma unroll
  for (int rnd = 0; rnd < 4; rnd++) {
    const int idx = t + rnd * 256;
    const int r = idx >> 3, cg = idx & 7;
    *(int4*)(&Qs[r * LQ + cg * 8]) =
        *(const int4*)(&qh[((size_t)h * NN + qBlk + r) * HWC + cg * 8]);
  }
#endif
  __syncthreads();

  // Q A-fragments (16 rows per wave, K=64 -> two 16x32 fragments), reused every tile
  Frag qf[2];
  {
    const int rb = wave * 16;
#pragma unroll
    for (int c = 0; c < 2; c++) {
      qf[c].q[0] = *(const int4*)(&Qs[(rb + lm) * LQ + c * 32 + hh * 8]);
      qf[c].q[1] = *(const int4*)(&Qs[(rb + lm) * LQ + c * 32 + 16 + hh * 8]);
    }
  }

  v8f O[4];
#pragma unroll
  for (int cc = 0; cc < 4; cc++) O[cc] = (v8f){};
  float mrow[8], lrow[8];
#pragma unroll
  for (int r = 0; r < 8; r++) { mrow[r] = -INFINITY; lrow[r] = 0.f; }

  for (int kt = 0; kt < NN / 32; kt++) {
    // stage K tile (32x64 row-major): TDM DMA; V^T tile (64x32): manual transpose
#if HAS_TDM
    if (wave == 0)
      tdm_load_2d_f16((unsigned)(uintptr_t)Ks,
                      &kh[((size_t)h * NN + kt * 32) * HWC], 64, 32, 64ull, 4, 3);
#endif
    {
      const int r = t >> 3, cg = t & 7;
      const size_t gro = ((size_t)h * NN + kt * 32 + r) * HWC + cg * 8;
#if !HAS_TDM
      *(int4*)(&Ks[r * LK + cg * 8]) = *(const int4*)(&kh[gro]);
#endif
      F16x8 vv; vv.i = *(const int4*)(&vh[gro]);
#pragma unroll
      for (int e = 0; e < 8; e++) VTs[(cg * 8 + e) * LV + r] = vv.h[e];
    }
#if HAS_TDM
    if (wave == 0) __builtin_amdgcn_s_wait_tensorcnt(0);
#endif
    __syncthreads();

    // S = q . k^T  (two 16-col halves, K=64 = 2 wmma each)
    v8f s[2];
#pragma unroll
    for (int kc = 0; kc < 2; kc++) {
      Frag b0, b1;
      const int rr = (kc * 16 + lm) * LK;
      b0.q[0] = *(const int4*)(&Ks[rr + hh * 8]);
      b0.q[1] = *(const int4*)(&Ks[rr + 16 + hh * 8]);
      b1.q[0] = *(const int4*)(&Ks[rr + 32 + hh * 8]);
      b1.q[1] = *(const int4*)(&Ks[rr + 32 + 16 + hh * 8]);
      v8f z = (v8f){};
      z = wmma_f16(qf[0].v, b0.v, z);
      z = wmma_f16(qf[1].v, b1.v, z);
      s[kc] = z;
    }

    // bias + mask in C layout
    const int mv0 = mask[kt * 32 + lm];
    const int mv1 = mask[kt * 32 + 16 + lm];
#pragma unroll
    for (int r = 0; r < 8; r++) {
      const int qrow = qBlk + wave * 16 + r + 8 * hh;
      const float* bp = &bias[((size_t)h * NN + qrow) * NN + kt * 32];
      s[0][r] = mv0 ? (s[0][r] + bp[lm]) : -INFINITY;
      s[1][r] = mv1 ? (s[1][r] + bp[16 + lm]) : -INFINITY;
    }
    if (kt + 1 < NN / 32)  // pull next bias tile toward L2 (global_prefetch_b8)
      __builtin_prefetch(
          &bias[((size_t)h * NN + qBlk + wave * 16 + hh * 8) * NN + (kt + 1) * 32 + lane],
          0, 0);

    // online softmax; row data for element r lives in 16 lanes of one half
#pragma unroll
    for (int r = 0; r < 8; r++) {
      float rm = redmax16(fmaxf(s[0][r], s[1][r]));
      const float mn = fmaxf(mrow[r], rm);
      float alpha, p0, p1;
      if (mn == -INFINITY) { alpha = 1.f; p0 = 0.f; p1 = 0.f; }
      else {
        alpha = __expf(mrow[r] - mn);
        p0 = __expf(s[0][r] - mn);
        p1 = __expf(s[1][r] - mn);
      }
      mrow[r] = mn;
      lrow[r] = lrow[r] * alpha + redsum16(p0 + p1);
#pragma unroll
      for (int cc = 0; cc < 4; cc++) O[cc][r] *= alpha;
      Pw[wave][(r + 8 * hh) * LP + lm]      = (_Float16)p0;
      Pw[wave][(r + 8 * hh) * LP + 16 + lm] = (_Float16)p1;
    }

    // relayout P (C->A) via per-wave LDS, then O += P . V
    Frag pf;
    pf.q[0] = *(const int4*)(&Pw[wave][lm * LP + hh * 8]);
    pf.q[1] = *(const int4*)(&Pw[wave][lm * LP + 16 + hh * 8]);
#pragma unroll
    for (int cc = 0; cc < 4; cc++) {
      Frag vf;
      const int rr = (cc * 16 + lm) * LV;
      vf.q[0] = *(const int4*)(&VTs[rr + hh * 8]);
      vf.q[1] = *(const int4*)(&VTs[rr + 16 + hh * 8]);
      O[cc] = wmma_f16(pf.v, vf.v, O[cc]);
    }
    __syncthreads();
  }

  // normalize and write y (N, H*HW) f16
#pragma unroll
  for (int r = 0; r < 8; r++) {
    const float inv = lrow[r] > 0.f ? 1.f / lrow[r] : 0.f;
    const int qrow = qBlk + wave * 16 + r + 8 * hh;
#pragma unroll
    for (int cc = 0; cc < 4; cc++)
      yh[(size_t)qrow * DD + h * HWC + cc * 16 + lm] = (_Float16)(O[cc][r] * inv);
  }
}

// ---------------- host launcher ----------------
extern "C" void kernel_launch(void* const* d_in, const int* in_sizes, int n_in,
                              void* d_out, int out_size, void* d_ws, size_t ws_size,
                              hipStream_t stream) {
  (void)in_sizes; (void)n_in; (void)out_size; (void)ws_size;
  const float* x    = (const float*)d_in[0];
  const float* bias = (const float*)d_in[1];
  const int*   mask = (const int*)d_in[2];
  const float* ln_w = (const float*)d_in[3];
  const float* ln_b = (const float*)d_in[4];
  const float* Wqkv = (const float*)d_in[5];
  const float* Wo   = (const float*)d_in[6];
  const float* bo   = (const float*)d_in[7];
  const float* Wg   = (const float*)d_in[8];
  const float* bg   = (const float*)d_in[9];
  float* out = (float*)d_out;

  char* ws = (char*)d_ws;
  size_t off = 0;
  auto take = [&](size_t nelem) -> _Float16* {
    _Float16* p = (_Float16*)(ws + off);
    off += nelem * sizeof(_Float16);
    return p;
  };
  _Float16* xn   = take((size_t)NN * DD);
  _Float16* wqkv = take((size_t)3 * DD * DD);
  _Float16* wg   = take((size_t)DD * DD);
  _Float16* wo   = take((size_t)DD * DD);
  _Float16* qh   = take((size_t)NH * NN * HWC);
  _Float16* kh   = take((size_t)NH * NN * HWC);
  _Float16* vh   = take((size_t)NH * NN * HWC);
  _Float16* yh   = take((size_t)NN * DD);
  _Float16* gy   = take((size_t)NN * DD);

  cvt_kernel<<<512, 256, 0, stream>>>(Wqkv, wqkv, 3 * DD * DD);
  cvt_kernel<<<512, 256, 0, stream>>>(Wg, wg, DD * DD);
  cvt_kernel<<<512, 256, 0, stream>>>(Wo, wo, DD * DD);
  ln_kernel<<<NN, 256, 0, stream>>>(x, ln_w, ln_b, xn);

  gemm_kernel<0><<<dim3(3 * DD / 128, NN / 128), 256, 0, stream>>>(
      xn, wqkv, DD, nullptr, nullptr, nullptr, nullptr, qh, kh, vh);

  attn_kernel<<<dim3(NN / 128, NH), 256, 0, stream>>>(qh, kh, vh, bias, mask, yh);

  gemm_kernel<1><<<dim3(DD / 128, NN / 128), 256, 0, stream>>>(
      xn, wg, DD, bg, yh, gy, nullptr, nullptr, nullptr, nullptr);

  gemm_kernel<2><<<dim3(DD / 128, NN / 128), 256, 0, stream>>>(
      gy, wo, DD, bo, nullptr, nullptr, out, nullptr, nullptr, nullptr);
}
